// MoELayer_58798102282706
// MI455X (gfx1250) — matrix operand order repaired and checked by reading, not compiled, not run on hardware
//
#include <hip/hip_runtime.h>
#include <math.h>

// Problem constants (B=2, S=2048 -> T=4096)
#define TT 4096
#define DD 1024
#define HH 4096
#define EE 8
#define RTOT (2*TT)          // total expert-row assignments (top-2, distinct experts)
#define NRB (TT/8)           // router blocks (8 tokens per 256-thread block)

// GEMM tiling
#define BM 128
#define BN 64
#define BK 32
#define LDT 40               // LDS row stride in halfwords (32 + 8 pad = 80B, 16B aligned)

typedef __attribute__((ext_vector_type(16))) __bf16 v16bf;
typedef __attribute__((ext_vector_type(8)))  float  v8f;
typedef __attribute__((ext_vector_type(4)))  unsigned int v4u;
typedef __attribute__((ext_vector_type(8)))  int v8i;
typedef __attribute__((ext_vector_type(4)))  int v4i;

union FragAB { uint4 u[2]; v16bf v; };
union FragC  { v8f v; float f[8]; };

__device__ __forceinline__ unsigned short f2bf(float f) {
  union { float f; unsigned int u; } c; c.f = f;
  unsigned int u = c.u + 0x7FFFu + ((c.u >> 16) & 1u);  // round-to-nearest-even
  return (unsigned short)(u >> 16);
}

__device__ __forceinline__ float fast_silu(float x) {
  // x * sigmoid(x); v_exp_f32 + v_rcp_f32, no IEEE-div expansion
  return x * __builtin_amdgcn_rcpf(1.f + __expf(-x));
}

// TDM: load a BN x BK (rows x elements) bf16 tile, row stride `row_stride_elems`,
// into LDS at `lds_addr` with 80B LDS row pitch (pad 4 DWORDs after every 64B row).
__device__ __forceinline__ void tdm_load_tile_b(unsigned lds_addr,
                                                const unsigned short* gsrc,
                                                unsigned row_stride_elems) {
  unsigned long long ga = (unsigned long long)(uintptr_t)gsrc;
  v4u g0 = { 1u,                                   // count=1 (valid user descriptor)
             lds_addr,                             // LDS byte address
             (unsigned)ga,                         // global_addr[31:0]
             (unsigned)((ga >> 32) & 0x01FFFFFFu) | 0x80000000u };  // addr[56:32] | type=2
  unsigned d0 = row_stride_elems;                  // tensor_dim0 (>= tile span, no OOB)
  unsigned d1 = 1u << 20;                          // tensor_dim1 (plenty)
  v8i g1;
  g1[0] = (int)((1u << 16)      // data_size = 1 -> 2 bytes
              | (1u << 20)      // pad_enable
              | (3u << 22)      // pad_interval: after 16 DWORDs (64B)
              | (3u << 25));    // pad_amount: 4 DWORDs (16B)
  g1[1] = (int)(d0 << 16);                         // tensor_dim0[15:0] (d0 < 65536)
  g1[2] = (int)((d0 >> 16) | (d1 << 16));          // dim0 hi | dim1 lo
  g1[3] = (int)((d1 >> 16) | ((unsigned)BK << 16));// dim1 hi | tile_dim0 = BK elems
  g1[4] = BN;                                      // tile_dim1 = BN rows, tile_dim2 = 0
  g1[5] = (int)d0;                                 // tensor_dim0_stride lo
  g1[6] = 0;                                       // stride0 hi | stride1 lo
  g1[7] = 0;
  v4i g2 = { 1, 1, 0, 0 };                         // tensor_dim2=1, tensor_dim3=1
  v4i g3 = { 0, 0, 0, 0 };
  v8i g4 = { 0, 0, 0, 0, 0, 0, 0, 0 };             // unused extra group (2D tile)
  __builtin_amdgcn_tensor_load_to_lds(g0, g1, g2, g3, g4, 0);
}

// ---------------------------------------------------------------- init
__global__ void k_init(int* cnt, int* cnt2) {
  if (threadIdx.x < EE) { cnt[threadIdx.x] = 0; cnt2[threadIdx.x] = 0; }
}

// ---------------------------------------------------------------- x -> bf16
__global__ __launch_bounds__(256) void k_xbf(const float* __restrict__ x,
                                             unsigned short* __restrict__ xbf) {
  int idx = blockIdx.x * 256 + threadIdx.x;          // over T*D/4
  float4 v = ((const float4*)x)[idx];
  ushort4 o;
  o.x = f2bf(v.x); o.y = f2bf(v.y); o.z = f2bf(v.z); o.w = f2bf(v.w);
  ((ushort4*)xbf)[idx] = o;
}

// ---------------------------------------------------------------- weight transpose + bf16
// W [e][K][N] fp32  ->  WT [e][N][K] bf16   (grid: (N/32, K/32, E), 256 thr)
__global__ __launch_bounds__(256) void k_wt(const float* __restrict__ W,
                                            unsigned short* __restrict__ WT,
                                            int K, int N) {
  __shared__ unsigned short t[32][33];
  int e = blockIdx.z;
  const float* src = W + (size_t)e * K * N;
  unsigned short* dst = WT + (size_t)e * K * N;
  int n0 = blockIdx.x * 32, k0 = blockIdx.y * 32;
  int tx = threadIdx.x & 31, ty = threadIdx.x >> 5;   // 8 rows per pass
#pragma unroll
  for (int i = 0; i < 32; i += 8)
    t[tx][ty + i] = f2bf(src[(size_t)(k0 + ty + i) * N + n0 + tx]);
  __syncthreads();
#pragma unroll
  for (int i = 0; i < 32; i += 8)
    dst[(size_t)(n0 + ty + i) * K + k0 + tx] = t[ty + i][tx];
}

// ---------------------------------------------------------------- router (one wave per token)
__global__ __launch_bounds__(256) void k_router(const float* __restrict__ x,
                                                const float* __restrict__ Wg,
                                                int* __restrict__ topi,
                                                float* __restrict__ topw,
                                                int* __restrict__ cnt,
                                                float* __restrict__ blockPsum) {
  __shared__ float sm[8][EE];
  int tid = threadIdx.x, lane = tid & 31, wave = tid >> 5;
  int t = blockIdx.x * 8 + wave;

  float acc[EE];
#pragma unroll
  for (int e = 0; e < EE; e++) acc[e] = 0.f;

  const float* xp = x + (size_t)t * DD;
  for (int i = 0; i < DD / 32; i++) {
    int d = i * 32 + lane;
    float xv = xp[d];
#pragma unroll
    for (int e = 0; e < EE; e++) acc[e] += xv * Wg[e * DD + d];
  }
#pragma unroll
  for (int e = 0; e < EE; e++) {
#pragma unroll
    for (int o = 16; o > 0; o >>= 1) acc[e] += __shfl_xor(acc[e], o, 32);
  }

  if (lane == 0) {
    float m = acc[0];
#pragma unroll
    for (int e = 1; e < EE; e++) m = fmaxf(m, acc[e]);
    float p[EE]; float s = 0.f;
#pragma unroll
    for (int e = 0; e < EE; e++) { p[e] = __expf(acc[e] - m); s += p[e]; }
    float inv = 1.f / s;
#pragma unroll
    for (int e = 0; e < EE; e++) { p[e] *= inv; sm[wave][e] = p[e]; }

    int i0 = 0;
#pragma unroll
    for (int e = 1; e < EE; e++) if (p[e] > p[i0]) i0 = e;   // lowest index on tie
    int i1 = (i0 == 0) ? 1 : 0;
#pragma unroll
    for (int e = 0; e < EE; e++) if (e != i0 && p[e] > p[i1]) i1 = e;

    float tw = p[i0] + p[i1];
    topi[2 * t]     = i0; topi[2 * t + 1] = i1;
    topw[2 * t]     = p[i0] / tw;
    topw[2 * t + 1] = p[i1] / tw;
    atomicAdd(&cnt[i0], 1);
    atomicAdd(&cnt[i1], 1);
  }
  __syncthreads();
  if (tid < EE) {                       // deterministic per-block prob partials
    float s = 0.f;
#pragma unroll
    for (int w = 0; w < 8; w++) s += sm[w][tid];
    blockPsum[blockIdx.x * EE + tid] = s;
  }
}

// ---------------------------------------------------------------- finalize (offsets + aux loss)
__global__ void k_finalize(const int* __restrict__ cnt,
                           const float* __restrict__ blockPsum,
                           int* __restrict__ offs, float* __restrict__ out_aux) {
  float ps[EE];
#pragma unroll
  for (int e = 0; e < EE; e++) ps[e] = 0.f;
  for (int b = 0; b < NRB; b++)
    for (int e = 0; e < EE; e++) ps[e] += blockPsum[b * EE + e];
  int o = 0;
  for (int e = 0; e < EE; e++) { offs[e] = o; o += cnt[e]; }
  offs[EE] = o;
  float a = 0.f;
  for (int e = 0; e < EE; e++) a += (float)cnt[e] * ps[e];
  *out_aux = (float)EE * a / ((float)TT * (float)TT);   // E * sum(f*p)
}

// ---------------------------------------------------------------- build compact per-expert rows
__global__ __launch_bounds__(256) void k_build(const int* __restrict__ topi,
                                               const float* __restrict__ topw,
                                               const int* __restrict__ offs,
                                               int* __restrict__ cnt2,
                                               int* __restrict__ rowTok,
                                               float* __restrict__ rowW,
                                               int* __restrict__ tokRow) {
  int t = blockIdx.x * 256 + threadIdx.x;
  if (t >= TT) return;
#pragma unroll
  for (int k = 0; k < 2; k++) {
    int e = topi[2 * t + k];
    int pos = atomicAdd(&cnt2[e], 1);
    int r = offs[e] + pos;
    rowTok[r] = t;
    rowW[r] = topw[2 * t + k];
    tokRow[2 * t + k] = r;
  }
}

// ---------------------------------------------------------------- GEMM1: H = silu(gather(x) @ W1[e] + b1[e])
// A: gathered bf16 token rows; B: W1T bf16 [e][n][k] via TDM; double-buffered LDS.
__global__ __launch_bounds__(256) void k_gemm1(const unsigned short* __restrict__ xbf,
                                               const unsigned short* __restrict__ W1T,
                                               const float* __restrict__ b1,
                                               const int* __restrict__ cnt,
                                               const int* __restrict__ offs,
                                               const int* __restrict__ rowTok,
                                               unsigned short* __restrict__ Hbuf) {
  int e = blockIdx.z;
  int Ne = cnt[e];
  int mbase = blockIdx.y * BM;
  if (mbase >= Ne) return;
  int nbase = blockIdx.x * BN;
  int off_e = offs[e];

  __shared__ unsigned short As[2][BM * LDT];
  __shared__ unsigned short Bs[2][BN * LDT];

  int tid = threadIdx.x, lane = tid & 31, wave = tid >> 5;
  int mwave = (wave >> 1) * 32, nwave = (wave & 1) * 32;
  int lr = lane & 15, hs = lane >> 4;

  const unsigned short* bsrc = W1T + ((size_t)e * HH + nbase) * DD;   // row n, stride DD

  FragC acc[2][2];
#pragma unroll
  for (int mi = 0; mi < 2; mi++)
#pragma unroll
    for (int ni = 0; ni < 2; ni++)
#pragma unroll
      for (int v = 0; v < 8; v++) acc[mi][ni].f[v] = 0.f;

  int ar = tid >> 1;                 // A-load duty: row 0..127
  int acb = (tid & 1) * 16;          // 16 halfwords per thread
  int atok = -1;
  if (mbase + ar < Ne) atok = rowTok[off_e + mbase + ar];
  const unsigned short* arow = (atok >= 0) ? (xbf + (size_t)atok * DD) : xbf;

  // ---- prologue: fill buffer 0
  if (wave == 0)
    tdm_load_tile_b((unsigned)(uintptr_t)&Bs[0][0], bsrc, DD);
  {
    uint4 a0 = {0, 0, 0, 0}, a1 = {0, 0, 0, 0};
    if (atok >= 0) {
      const uint4* p = (const uint4*)(arow + acb);
      a0 = p[0]; a1 = p[1];
    }
    *(uint4*)&As[0][ar * LDT + acb]     = a0;
    *(uint4*)&As[0][ar * LDT + acb + 8] = a1;
  }
  if (wave == 0) __builtin_amdgcn_s_wait_tensorcnt(0);
  __syncthreads();

  const int NT = DD / BK;
  for (int it = 0; it < NT; ++it) {
    int c = it & 1;
    bool nxt = (it + 1 < NT);

    // ---- issue next tile (overlaps with WMMA below)
    if (nxt && wave == 0)
      tdm_load_tile_b((unsigned)(uintptr_t)&Bs[1 - c][0], bsrc + (it + 1) * BK, DD);
    uint4 a0 = {0, 0, 0, 0}, a1 = {0, 0, 0, 0};
    if (nxt && atok >= 0) {
      const uint4* p = (const uint4*)(arow + (it + 1) * BK + acb);
      a0 = p[0]; a1 = p[1];
    }

    // ---- compute current tile
    FragAB fa[2], fb[2];
#pragma unroll
    for (int mi = 0; mi < 2; mi++) {
      const unsigned short* ap = &As[c][(mwave + mi * 16 + lr) * LDT];
      fa[mi].u[0] = *(const uint4*)(ap + hs * 8);        // K = 8h..8h+7
      fa[mi].u[1] = *(const uint4*)(ap + 16 + hs * 8);   // K = 16+8h..23+8h
    }
#pragma unroll
    for (int ni = 0; ni < 2; ni++) {
      const unsigned short* bp = &Bs[c][(nwave + ni * 16 + lr) * LDT];
      fb[ni].u[0] = *(const uint4*)(bp + hs * 8);        // K = 8h..8h+7
      fb[ni].u[1] = *(const uint4*)(bp + 16 + hs * 8);   // K = 16+8h..23+8h
    }
#pragma unroll
    for (int mi = 0; mi < 2; mi++)
#pragma unroll
      for (int ni = 0; ni < 2; ni++)
        acc[mi][ni].v = __builtin_amdgcn_wmma_f32_16x16x32_bf16(
            false, fa[mi].v, false, fb[ni].v, (short)0, acc[mi][ni].v, false, false);

    // ---- stage next A into the other buffer
    if (nxt) {
      *(uint4*)&As[1 - c][ar * LDT + acb]     = a0;
      *(uint4*)&As[1 - c][ar * LDT + acb + 8] = a1;
    }
    if (wave == 0) __builtin_amdgcn_s_wait_tensorcnt(0);
    __syncthreads();
  }

#pragma unroll
  for (int mi = 0; mi < 2; mi++)
#pragma unroll
    for (int ni = 0; ni < 2; ni++) {
      int col = nbase + nwave + ni * 16 + lr;
      float bias = b1[(size_t)e * HH + col];
#pragma unroll
      for (int v = 0; v < 8; v++) {
        int grow = mbase + mwave + mi * 16 + v + 8 * hs;
        if (grow < Ne) {
          float hv = fast_silu(acc[mi][ni].f[v] + bias);
          Hbuf[(size_t)(off_e + grow) * HH + col] = f2bf(hv);
        }
      }
    }
}

// ---------------------------------------------------------------- GEMM2: Y = w * (H @ W2[e] + b2[e])
__global__ __launch_bounds__(256) void k_gemm2(const unsigned short* __restrict__ Hbuf,
                                               const unsigned short* __restrict__ W2T,
                                               const float* __restrict__ b2,
                                               const int* __restrict__ cnt,
                                               const int* __restrict__ offs,
                                               const float* __restrict__ rowW,
                                               float* __restrict__ Ybuf) {
  int e = blockIdx.z;
  int Ne = cnt[e];
  int mbase = blockIdx.y * BM;
  if (mbase >= Ne) return;
  int nbase = blockIdx.x * BN;
  int off_e = offs[e];

  __shared__ unsigned short As[2][BM * LDT];
  __shared__ unsigned short Bs[2][BN * LDT];

  int tid = threadIdx.x, lane = tid & 31, wave = tid >> 5;
  int mwave = (wave >> 1) * 32, nwave = (wave & 1) * 32;
  int lr = lane & 15, hs = lane >> 4;

  const unsigned short* bsrc = W2T + ((size_t)e * DD + nbase) * HH;   // row n, stride HH

  FragC acc[2][2];
#pragma unroll
  for (int mi = 0; mi < 2; mi++)
#pragma unroll
    for (int ni = 0; ni < 2; ni++)
#pragma unroll
      for (int v = 0; v < 8; v++) acc[mi][ni].f[v] = 0.f;

  int ar = tid >> 1;
  int acb = (tid & 1) * 16;
  bool avalid = (mbase + ar < Ne);
  const unsigned short* arow = Hbuf + (size_t)(off_e + mbase + ar) * HH;

  if (wave == 0)
    tdm_load_tile_b((unsigned)(uintptr_t)&Bs[0][0], bsrc, HH);
  {
    uint4 a0 = {0, 0, 0, 0}, a1 = {0, 0, 0, 0};
    if (avalid) {
      const uint4* p = (const uint4*)(arow + acb);
      a0 = p[0]; a1 = p[1];
    }
    *(uint4*)&As[0][ar * LDT + acb]     = a0;
    *(uint4*)&As[0][ar * LDT + acb + 8] = a1;
  }
  if (wave == 0) __builtin_amdgcn_s_wait_tensorcnt(0);
  __syncthreads();

  const int NT = HH / BK;
  for (int it = 0; it < NT; ++it) {
    int c = it & 1;
    bool nxt = (it + 1 < NT);

    if (nxt && wave == 0)
      tdm_load_tile_b((unsigned)(uintptr_t)&Bs[1 - c][0], bsrc + (it + 1) * BK, HH);
    uint4 a0 = {0, 0, 0, 0}, a1 = {0, 0, 0, 0};
    if (nxt && avalid) {
      const uint4* p = (const uint4*)(arow + (it + 1) * BK + acb);
      a0 = p[0]; a1 = p[1];
    }

    FragAB fa[2], fb[2];
#pragma unroll
    for (int mi = 0; mi < 2; mi++) {
      const unsigned short* ap = &As[c][(mwave + mi * 16 + lr) * LDT];
      fa[mi].u[0] = *(const uint4*)(ap + hs * 8);
      fa[mi].u[1] = *(const uint4*)(ap + 16 + hs * 8);
    }
#pragma unroll
    for (int ni = 0; ni < 2; ni++) {
      const unsigned short* bp = &Bs[c][(nwave + ni * 16 + lr) * LDT];
      fb[ni].u[0] = *(const uint4*)(bp + hs * 8);
      fb[ni].u[1] = *(const uint4*)(bp + 16 + hs * 8);
    }
#pragma unroll
    for (int mi = 0; mi < 2; mi++)
#pragma unroll
      for (int ni = 0; ni < 2; ni++)
        acc[mi][ni].v = __builtin_amdgcn_wmma_f32_16x16x32_bf16(
            false, fa[mi].v, false, fb[ni].v, (short)0, acc[mi][ni].v, false, false);

    if (nxt) {
      *(uint4*)&As[1 - c][ar * LDT + acb]     = a0;
      *(uint4*)&As[1 - c][ar * LDT + acb + 8] = a1;
    }
    if (wave == 0) __builtin_amdgcn_s_wait_tensorcnt(0);
    __syncthreads();
  }

#pragma unroll
  for (int mi = 0; mi < 2; mi++)
#pragma unroll
    for (int ni = 0; ni < 2; ni++) {
      int col = nbase + nwave + ni * 16 + lr;
      float bias = b2[(size_t)e * DD + col];
#pragma unroll
      for (int v = 0; v < 8; v++) {
        int grow = mbase + mwave + mi * 16 + v + 8 * hs;
        if (grow < Ne) {
          int cr = off_e + grow;
          Ybuf[(size_t)cr * DD + col] = rowW[cr] * (acc[mi][ni].f[v] + bias);
        }
      }
    }
}

// ---------------------------------------------------------------- combine: out[t] = Y[row0] + Y[row1]
__global__ __launch_bounds__(256) void k_combine(const float* __restrict__ Ybuf,
                                                 const int* __restrict__ tokRow,
                                                 float* __restrict__ out) {
  int t = blockIdx.x;
  int c = threadIdx.x;                      // 256 threads * float4 = 1024 = D
  int r0 = tokRow[2 * t], r1 = tokRow[2 * t + 1];
  float4 a = ((const float4*)(Ybuf + (size_t)r0 * DD))[c];
  float4 b = ((const float4*)(Ybuf + (size_t)r1 * DD))[c];
  float4 o = { a.x + b.x, a.y + b.y, a.z + b.z, a.w + b.w };
  ((float4*)(out + (size_t)t * DD))[c] = o;
}

// ---------------------------------------------------------------- launch
extern "C" void kernel_launch(void* const* d_in, const int* in_sizes, int n_in,
                              void* d_out, int out_size, void* d_ws, size_t ws_size,
                              hipStream_t stream) {
  (void)in_sizes; (void)n_in; (void)out_size; (void)ws_size;
  const float* x  = (const float*)d_in[0];
  const float* Wg = (const float*)d_in[1];
  const float* W1 = (const float*)d_in[2];
  const float* b1 = (const float*)d_in[3];
  const float* W2 = (const float*)d_in[4];
  const float* b2 = (const float*)d_in[5];
  float* out = (float*)d_out;

  char* ws = (char*)d_ws;
  size_t o = 0;
  unsigned short* xbf  = (unsigned short*)(ws + o); o += (size_t)TT * DD * 2;   // 8 MB
  unsigned short* W1T  = (unsigned short*)(ws + o); o += (size_t)EE * DD * HH * 2; // 64 MB
  unsigned short* W2T  = (unsigned short*)(ws + o); o += (size_t)EE * HH * DD * 2; // 64 MB
  unsigned short* Hbuf = (unsigned short*)(ws + o); o += (size_t)RTOT * HH * 2; // 64 MB
  float* Ybuf   = (float*)(ws + o); o += (size_t)RTOT * DD * 4;                 // 32 MB
  int*   rowTok = (int*)(ws + o);   o += (size_t)RTOT * 4;
  float* rowW   = (float*)(ws + o); o += (size_t)RTOT * 4;
  int*   topi   = (int*)(ws + o);   o += (size_t)RTOT * 4;
  float* topw   = (float*)(ws + o); o += (size_t)RTOT * 4;
  int*   tokRow = (int*)(ws + o);   o += (size_t)RTOT * 4;
  float* blockPsum = (float*)(ws + o); o += (size_t)NRB * EE * 4;
  int*   cnt  = (int*)(ws + o); o += 16 * 4;
  int*   cnt2 = (int*)(ws + o); o += 16 * 4;
  int*   offs = (int*)(ws + o); o += 16 * 4;

  k_init    <<<1, 32, 0, stream>>>(cnt, cnt2);
  k_xbf     <<<(TT * DD / 4) / 256, 256, 0, stream>>>(x, xbf);
  k_wt      <<<dim3(HH / 32, DD / 32, EE), 256, 0, stream>>>(W1, W1T, DD, HH);
  k_wt      <<<dim3(DD / 32, HH / 32, EE), 256, 0, stream>>>(W2, W2T, HH, DD);
  k_router  <<<NRB, 256, 0, stream>>>(x, Wg, topi, topw, cnt, blockPsum);
  k_finalize<<<1, 1, 0, stream>>>(cnt, blockPsum, offs, out + (size_t)TT * DD);
  k_build   <<<TT / 256, 256, 0, stream>>>(topi, topw, offs, cnt2, rowTok, rowW, tokRow);
  k_gemm1   <<<dim3(HH / BN, TT / BM, EE), 256, 0, stream>>>(xbf, W1T, b1, cnt, offs, rowTok, Hbuf);
  k_gemm2   <<<dim3(DD / BN, TT / BM, EE), 256, 0, stream>>>(Hbuf, W2T, b2, cnt, offs, rowW, Ybuf);
  k_combine <<<TT, 256, 0, stream>>>(Ybuf, tokRow, out);
}
